// AutoregressiveSubsetSampler_66924180406433
// MI455X (gfx1250) — matrix operand order repaired and checked by reading, not compiled, not run on hardware
//
#include <hip/hip_runtime.h>
#include <stdint.h>

// ---------------------------------------------------------------------------
// Autoregressive subset sampler for MI455X (gfx1250, wave32).
//
// Sequential scan (6144 steps, 256-dim carry) => latency problem for ONE
// persistent workgroup (32 wave32 waves on one WGP).  Weights are pre-
// converted to f16 and pre-swizzled into the v_wmma_f32_16x16x32_f16
// B-operand lane layout (one contiguous 32B load per lane per 32x16 tile).
// Wv/Wo (256 KB) are staged ONCE into dynamic LDS (CDNA5: 320 KB/WGP) so the
// attention matvecs run entirely out of ds_load_b128 — no L2 traffic and no
// VGPR/scratch spills from the compiler trying to keep them register-
// resident.  Wf1/Wf2 (2 MB) stream from L2 with a double-buffered B pipeline
// (load kt+1 while WMMA consumes kt => waits on loadcnt<=1, not 0).
// Matvecs are WMMA with x replicated over the 16 A rows; C row M=0 (VGPR0,
// lanes 0-15) is the result.  The 8-try rejection sampler early-exits
// block-wide (tries k and k+4 share one threefry call).  All reductions are
// fixed-order trees -> bitwise deterministic across calls.
// ---------------------------------------------------------------------------

typedef __attribute__((ext_vector_type(16))) _Float16 v16h;
typedef __attribute__((ext_vector_type(8)))  _Float16 v8h;
typedef __attribute__((ext_vector_type(8)))  float    v8f;

#define DDIM     256
#define NCOLS    4096
#define FFDIM    2048
#define NNODES   4096
#define TSTEPS   6144      // NNODES * 1.5
#define NTHREADS 1024
#define NWAVES   32
#define WHALF    65536     // halfs per 256x256 swizzled weight
#define DYN_LDS  (2 * WHALF * sizeof(_Float16))   // 256 KB for Wv+Wo

// ---------------- threefry2x32-20 (JAX-style counter PRNG) -----------------
__device__ __forceinline__ void threefry2x32(uint32_t k0, uint32_t k1,
                                             uint32_t x0, uint32_t x1,
                                             uint32_t& o0, uint32_t& o1) {
  uint32_t ks2 = 0x1BD11BDAu ^ k0 ^ k1;
  x0 += k0; x1 += k1;
#define TF_R(r) { x0 += x1; x1 = (x1 << (r)) | (x1 >> (32 - (r))); x1 ^= x0; }
  TF_R(13) TF_R(15) TF_R(26) TF_R(6)  x0 += k1;  x1 += ks2 + 1u;
  TF_R(17) TF_R(29) TF_R(16) TF_R(24) x0 += ks2; x1 += k0 + 2u;
  TF_R(13) TF_R(15) TF_R(26) TF_R(6)  x0 += k0;  x1 += k1 + 3u;
  TF_R(17) TF_R(29) TF_R(16) TF_R(24) x0 += k1;  x1 += ks2 + 4u;
  TF_R(13) TF_R(15) TF_R(26) TF_R(6)  x0 += ks2; x1 += k0 + 5u;
#undef TF_R
  o0 = x0; o1 = x1;
}

__device__ __forceinline__ float u01(uint32_t b) {
  return __uint_as_float(0x3f800000u | (b >> 9)) - 1.0f;
}

// ------------- weight pre-swizzle: f32 row-major -> f16 WMMA B tiles -------
// B element (k_local, n_local) of a 32x16 tile lives at lane = (k/16)*16 + n,
// half j = k % 16.  Flat: dst[((nt*KT+kt)*32 + lane)*16 + j].
__global__ void swz_kernel(const float* __restrict__ W,
                           _Float16* __restrict__ dst, int K, int N) {
  int KT = K / 32;
  size_t total = (size_t)K * N;
  for (size_t e = (size_t)blockIdx.x * blockDim.x + threadIdx.x; e < total;
       e += (size_t)gridDim.x * blockDim.x) {
    int tile = (int)(e >> 9);
    int r    = (int)(e & 511);
    int lane = r >> 4, j = r & 15;
    int nt = tile / KT, kt = tile % KT;
    int sk = kt * 32 + (lane >> 4) * 16 + j;
    int sn = nt * 16 + (lane & 15);
    dst[e] = (_Float16)W[(size_t)sk * N + sn];
  }
}

// ------------- matvec via WMMA: out = x(1xK) @ W(KxN), x replicated --------
// Compile-time shapes -> constant trip counts.  B double-buffered with a
// clamped next index (scalar select; no EXEC dance).  PREF=false for LDS-
// resident weights (flat prefetch of LDS lanes is a no-op per the ISA).
template <int KT, int NT, int KP, bool PREF>
__device__ __forceinline__ void matvec_wmma(const _Float16* __restrict__ Wt,
                                            const _Float16* xh, float* out,
                                            int wave, int lane) {
  constexpr int KTPER = KT / KP;            // k-tiles per work item
  constexpr int RITER = (NT * KP) / NWAVES; // work items per wave
  const int off = (lane & 16) ? 16 : 0;     // A-operand K phase per lane group
#pragma unroll 1
  for (int r = 0; r < RITER; ++r) {
    const int it = wave + r * NWAVES;
    const int nt = it % NT;
    const int kp = it / NT;
    const int k0 = kp * KTPER;
    const _Float16* tb = Wt + (size_t)nt * KT * 512 + (size_t)lane * 16;
    if constexpr (PREF) {
      // prefetch first tile of the next work item (clamped -> always valid)
      const int itn = (r + 1 < RITER) ? it + NWAVES : it;
      const int ntn = itn % NT, kpn = itn / NT;
      __builtin_prefetch(Wt + ((size_t)ntn * KT + kpn * KTPER) * 512 +
                             (size_t)lane * 16, 0, 2);
    }
    v8f c = {};
    v16h bcur = *(const v16h*)(tb + (size_t)k0 * 512);
#pragma unroll 4
    for (int kt = k0; kt < k0 + KTPER; ++kt) {
      const int ktn = (kt + 1 < k0 + KTPER) ? kt + 1 : kt;   // clamp
      v16h bnext = *(const v16h*)(tb + (size_t)ktn * 512);
      // A (16x32 f16, all M rows = x chunk): two aligned b128 LDS loads.
      union { v16h v; v8h h8[2]; } a;
      a.h8[0] = *(const v8h*)(xh + kt * 32 + off);
      a.h8[1] = *(const v8h*)(xh + kt * 32 + off + 8);
      c = __builtin_amdgcn_wmma_f32_16x16x32_f16(false, a.v, false, bcur,
                                                 (short)0, c, false, false);
      bcur = bnext;
    }
    if (lane < 16) out[it * 16 + lane] = c[0];  // C row M=0: VGPR0, lanes 0-15
  }
}

// --------------------------- persistent scan kernel ------------------------
__global__ void __launch_bounds__(NTHREADS)
scan_kernel(const float* __restrict__ ge,   const float* __restrict__ W1,
            const float* __restrict__ b1,   const float* __restrict__ W2,
            const float* __restrict__ b2,   const float* __restrict__ Wl,
            const float* __restrict__ bl,   const float* __restrict__ Wrow,
            const float* __restrict__ Wh,   const float* __restrict__ bgv,
            const float* __restrict__ bv,   const float* __restrict__ bo,
            const float* __restrict__ ln1g, const float* __restrict__ ln1b,
            const float* __restrict__ bf1,  const float* __restrict__ bf2,
            const float* __restrict__ ln2g, const float* __restrict__ ln2b,
            const _Float16* __restrict__ Wvh,  const _Float16* __restrict__ Woh,
            const _Float16* __restrict__ Wf1h, const _Float16* __restrict__ Wf2h,
            float* __restrict__ out) {
  __shared__ float s_g[DDIM];                    // scan carry
  __shared__ float s_xa[DDIM];                   // g_new / residual temp
  __shared__ float s_x1[DDIM];                   // post-LN1 activation
  __shared__ __align__(16) _Float16 s_xh[FFDIM]; // f16 matvec input
  __shared__ float s_mv[FFDIM];                  // matvec partial outputs
  __shared__ float s_redA[272], s_redB[272];     // fixed-order reductions
  __shared__ uint32_t s_bits[8 * 128];           // 8 tries x 4096 bits
  __shared__ int s_cnt[8];
  __shared__ int s_sel, s_done, s_nz;
  __shared__ int s_nzidx[64];
  __shared__ float s_h[2];
  __shared__ float s_mu, s_rstd;
  // 256 KB dynamic LDS: [0,WHALF) = Wv tiles, [WHALF,2*WHALF) = Wo tiles
  extern __shared__ __align__(32) _Float16 s_wdyn[];

  const int tid  = threadIdx.x;
  const int wave = tid >> 5, lane = tid & 31;

  // ---- one-time: stage Wv/Wo swizzled tiles into LDS (step-invariant) ----
  {
    const uint4* sv = (const uint4*)Wvh;
    const uint4* so = (const uint4*)Woh;
    uint4* dd = (uint4*)s_wdyn;
    for (int idx = tid; idx < WHALF / 8; idx += NTHREADS) {  // 8192 x 16B
      dd[idx] = sv[idx];
      dd[WHALF / 8 + idx] = so[idx];
    }
  }
  if (tid < DDIM) s_g[tid] = 0.0f;
  __syncthreads();

  for (int i = 0; i < TSTEPS; ++i) {
    // ---- per-step reset ----
    s_bits[tid] = 0u;                      // 1024 words == 8*128
    if (tid < 8) s_cnt[tid] = 0;
    if (tid == 0) { s_sel = 0; s_done = 0; s_nz = 0; }
    __syncthreads();

    // ---- h = relu([m,g] @ W1 + b1) @ W2 + b2 ----
    if (tid < DDIM) {
      float m  = (i < NNODES) ? ge[(size_t)i * DDIM + tid] : 0.0f;
      float gv = s_g[tid];
      s_redA[tid] = m * W1[tid * 2 + 0] + gv * W1[(DDIM + tid) * 2 + 0];
      s_redB[tid] = m * W1[tid * 2 + 1] + gv * W1[(DDIM + tid) * 2 + 1];
    }
    __syncthreads();
    if (tid < 16) {
      float a = 0.f, b = 0.f;
      for (int j = 0; j < 16; ++j) { a += s_redA[tid * 16 + j]; b += s_redB[tid * 16 + j]; }
      s_redA[256 + tid] = a; s_redB[256 + tid] = b;
    }
    __syncthreads();
    if (tid == 0) {
      float a = 0.f, b = 0.f;
      for (int j = 0; j < 16; ++j) { a += s_redA[256 + j]; b += s_redB[256 + j]; }
      float hp0 = fmaxf(a + b1[0], 0.f), hp1 = fmaxf(b + b1[1], 0.f);
      s_h[0] = hp0 * W2[0] + hp1 * W2[2] + b2[0];
      s_h[1] = hp0 * W2[1] + hp1 * W2[3] + b2[1];
    }
    __syncthreads();
    const float h0 = s_h[0], h1 = s_h[1];

    // ---- per-column Bernoulli probs (4 cols/thread, kept in regs) ----
    float p4[4];
    for (int q = 0; q < 4; ++q) {
      int c = tid + q * NTHREADS;
      float z = h0 * Wl[c] + h1 * Wl[NCOLS + c] + bl[c];
      p4[q] = 1.0f / (1.0f + __expf(-z));
    }

    // step key = fold_in(key(42), i)
    uint32_t k0, k1;
    threefry2x32(0u, 42u, 0u, (uint32_t)i, k0, k1);

    // ---- rejection loop, block-wide early exit.  One threefry call per col
    //      yields tries kk (out0) and kk+4 (out1); typically only kk=0 runs.
    for (int kk = 0; kk < 4; ++kk) {
      if (!s_done) {
        for (int q = 0; q < 4; ++q) {
          int c = tid + q * NTHREADS;
          uint32_t idx = (uint32_t)(kk * NCOLS + c);           // < 16384
          uint32_t r0, r1;
          threefry2x32(k0, k1, idx, idx + 16384u, r0, r1);
          uint32_t m0 = (u01(r0) < p4[q]) ? (1u << (c & 31)) : 0u;
          uint32_t m1 = (u01(r1) < p4[q]) ? (1u << (c & 31)) : 0u;
          if (m0) atomicOr(&s_bits[kk * 128 + (c >> 5)], m0);
          if (m1) atomicOr(&s_bits[(kk + 4) * 128 + (c >> 5)], m1);
        }
      }
      __syncthreads();
      if (!s_done && tid < 128)
        atomicAdd(&s_cnt[kk], (int)__popc(s_bits[kk * 128 + tid]));
      __syncthreads();
      if (tid == 0 && !s_done) {
        int cnt = s_cnt[kk];
        if (cnt == 0 || (cnt >= 2 && cnt <= 6)) { s_sel = kk; s_done = 1; }
      }
      __syncthreads();
    }
    if (!s_done) {  // uniform branch: check tries 4..7 (bits already made)
      if (tid < 512)
        atomicAdd(&s_cnt[4 + (tid >> 7)],
                  (int)__popc(s_bits[(4 + (tid >> 7)) * 128 + (tid & 127)]));
      __syncthreads();
      if (tid == 0) {
        int sel = 0;
        for (int k = 4; k < 8; ++k) {
          int cnt = s_cnt[k];
          if (cnt == 0 || (cnt >= 2 && cnt <= 6)) { sel = k; break; }
        }
        s_sel = sel;                       // argmax(all false) == 0 fallback
      }
      __syncthreads();
    }
    const int sel = s_sel;

    // ---- emit incidence row + deterministic nonzero-index gather ----
    for (int q = 0; q < 4; ++q) {
      int c = tid + q * NTHREADS;
      uint32_t bit = (s_bits[sel * 128 + (c >> 5)] >> (c & 31)) & 1u;
      out[(size_t)i * NCOLS + c] = (float)bit;
    }
    if (tid == 0) {
      int n = 0;
      for (int w = 0; w < 128 && n < 64; ++w) {
        uint32_t word = s_bits[sel * 128 + w];
        while (word && n < 64) {
          int b = __ffs(word) - 1;
          s_nzidx[n++] = w * 32 + b;
          word &= word - 1u;
        }
      }
      s_nz = n;
    }
    __syncthreads();

    // ---- g_new = tanh(row @ Wrow + h @ Wh + bg)  (sparse gather) ----
    if (tid < DDIM) {
      float acc = h0 * Wh[tid] + h1 * Wh[DDIM + tid] + bgv[tid];
      int n = s_nz;
      for (int t = 0; t < n; ++t) acc += Wrow[(size_t)s_nzidx[t] * DDIM + tid];
      float gn = tanhf(acc);
      s_xa[tid] = gn;
      s_xh[tid] = (_Float16)gn;
    }
    __syncthreads();

    // ---- attn = (g_new @ Wv + bv) @ Wo + bo   (B from LDS) ----
    matvec_wmma<8, 16, 2, false>(s_wdyn, s_xh, s_mv, wave, lane);
    __syncthreads();
    if (tid < DDIM) s_xh[tid] = (_Float16)(s_mv[tid] + s_mv[DDIM + tid] + bv[tid]);
    __syncthreads();
    matvec_wmma<8, 16, 2, false>(s_wdyn + WHALF, s_xh, s_mv, wave, lane);
    __syncthreads();

    // ---- x1 = LayerNorm(g_new + attn) ----
    if (tid < DDIM) {
      float v = s_xa[tid] + (s_mv[tid] + s_mv[DDIM + tid] + bo[tid]);
      s_x1[tid] = v;
      s_redA[tid] = v; s_redB[tid] = v * v;
    }
    __syncthreads();
    if (tid < 16) {
      float a = 0.f, b = 0.f;
      for (int j = 0; j < 16; ++j) { a += s_redA[tid * 16 + j]; b += s_redB[tid * 16 + j]; }
      s_redA[256 + tid] = a; s_redB[256 + tid] = b;
    }
    __syncthreads();
    if (tid == 0) {
      float a = 0.f, b = 0.f;
      for (int j = 0; j < 16; ++j) { a += s_redA[256 + j]; b += s_redB[256 + j]; }
      float mu = a / DDIM, var = b / DDIM - mu * mu;
      s_mu = mu; s_rstd = __frsqrt_rn(var + 1e-5f);
    }
    __syncthreads();
    if (tid < DDIM) {
      float v = ln1g[tid] * (s_x1[tid] - s_mu) * s_rstd + ln1b[tid];
      s_x1[tid] = v;
      s_xh[tid] = (_Float16)v;
    }
    __syncthreads();

    // ---- ff = relu(x1 @ Wf1 + bf1) @ Wf2 + bf2   (B streamed from L2) ----
    matvec_wmma<8, 128, 1, true>(Wf1h, s_xh, s_mv, wave, lane);
    __syncthreads();
    {
      int c0 = tid, c1 = tid + NTHREADS;
      s_xh[c0] = (_Float16)fmaxf(s_mv[c0] + bf1[c0], 0.f);
      s_xh[c1] = (_Float16)fmaxf(s_mv[c1] + bf1[c1], 0.f);
    }
    __syncthreads();
    matvec_wmma<64, 16, 2, true>(Wf2h, s_xh, s_mv, wave, lane);
    __syncthreads();

    // ---- g = LayerNorm(x1 + ff)  (scan carry) ----
    if (tid < DDIM) {
      float v = s_x1[tid] + (s_mv[tid] + s_mv[DDIM + tid] + bf2[tid]);
      s_xa[tid] = v;
      s_redA[tid] = v; s_redB[tid] = v * v;
    }
    __syncthreads();
    if (tid < 16) {
      float a = 0.f, b = 0.f;
      for (int j = 0; j < 16; ++j) { a += s_redA[tid * 16 + j]; b += s_redB[tid * 16 + j]; }
      s_redA[256 + tid] = a; s_redB[256 + tid] = b;
    }
    __syncthreads();
    if (tid == 0) {
      float a = 0.f, b = 0.f;
      for (int j = 0; j < 16; ++j) { a += s_redA[256 + j]; b += s_redB[256 + j]; }
      float mu = a / DDIM, var = b / DDIM - mu * mu;
      s_mu = mu; s_rstd = __frsqrt_rn(var + 1e-5f);
    }
    __syncthreads();
    if (tid < DDIM)
      s_g[tid] = ln2g[tid] * (s_xa[tid] - s_mu) * s_rstd + ln2b[tid];
    __syncthreads();
  }
}

// ---------------------------------------------------------------------------
extern "C" void kernel_launch(void* const* d_in, const int* in_sizes, int n_in,
                              void* d_out, int out_size, void* d_ws, size_t ws_size,
                              hipStream_t stream) {
  (void)in_sizes; (void)n_in; (void)out_size; (void)ws_size;

  const float* ge   = (const float*)d_in[0];
  // d_in[1] = b (unused by reference math)
  const float* W1   = (const float*)d_in[2];
  const float* b1   = (const float*)d_in[3];
  const float* W2   = (const float*)d_in[4];
  const float* b2   = (const float*)d_in[5];
  const float* Wl   = (const float*)d_in[6];
  const float* bl   = (const float*)d_in[7];
  const float* Wrow = (const float*)d_in[8];
  const float* Wh   = (const float*)d_in[9];
  const float* bg   = (const float*)d_in[10];
  const float* Wv   = (const float*)d_in[11];
  const float* bv   = (const float*)d_in[12];
  const float* Wo   = (const float*)d_in[13];
  const float* bo   = (const float*)d_in[14];
  const float* ln1g = (const float*)d_in[15];
  const float* ln1b = (const float*)d_in[16];
  const float* Wf1  = (const float*)d_in[17];
  const float* bf1  = (const float*)d_in[18];
  const float* Wf2  = (const float*)d_in[19];
  const float* bf2  = (const float*)d_in[20];
  const float* ln2g = (const float*)d_in[21];
  const float* ln2b = (const float*)d_in[22];

  // f16 swizzled-weight workspace: 2.25 MB of d_ws
  _Float16* wsh  = (_Float16*)d_ws;
  _Float16* Wvh  = wsh;                  //  65536 halfs
  _Float16* Woh  = wsh + 65536;          //  65536 halfs
  _Float16* Wf1h = wsh + 131072;         // 524288 halfs
  _Float16* Wf2h = wsh + 655360;         // 524288 halfs

  swz_kernel<<<256,  256, 0, stream>>>(Wv,  Wvh,  256,  256);
  swz_kernel<<<256,  256, 0, stream>>>(Wo,  Woh,  256,  256);
  swz_kernel<<<1024, 256, 0, stream>>>(Wf1, Wf1h, 256,  2048);
  swz_kernel<<<1024, 256, 0, stream>>>(Wf2, Wf2h, 2048, 256);

  scan_kernel<<<1, NTHREADS, DYN_LDS, stream>>>(
      ge, W1, b1, W2, b2, Wl, bl, Wrow, Wh, bg,
      bv, bo, ln1g, ln1b, bf1, bf2, ln2g, ln2b,
      Wvh, Woh, Wf1h, Wf2h, (float*)d_out);
}
